// MoEMlp_48722109005965
// MI455X (gfx1250) — matrix operand order repaired and checked by reading, not compiled, not run on hardware
//
#include <hip/hip_runtime.h>
#include <hip/hip_bf16.h>

#define CIN   256
#define HID   1024
#define NE    8
#define TM    128            // tokens per workgroup tile
#define HS    (HID + 8)      // H row stride in bf16 units (bank-conflict pad)
#define XS    (CIN + 12)     // x row stride in f32 units (bank-conflict pad)
#define NWAVE 8
#define USE_ASYNC 1

typedef __attribute__((ext_vector_type(16))) __bf16 v16bf;
typedef __attribute__((ext_vector_type(8)))  float  v8f;

union BFV {
    unsigned short s[16];
    uint4          u[2];
    v16bf          v;
};

static __device__ __forceinline__ unsigned short f2bf(float f) {
    unsigned int u = __float_as_uint(f);
    u += 0x7FFFu + ((u >> 16) & 1u);     // round-to-nearest-even
    return (unsigned short)(u >> 16);
}

// Branch-free GELU (tanh form). v_tanh_f32 is a TRANS32 op on gfx1250 and
// co-executes with WMMA; the exact-erf version expands to a divergent
// polynomial with exec-mask save/restore inside the hot loop.
static __device__ __forceinline__ float gelu_act(float h) {
    float u = 0.7978845608028654f * (h + 0.044715f * h * h * h);
#if __has_builtin(__builtin_amdgcn_tanhf)
    float t = __builtin_amdgcn_tanhf(u);
#else
    u = fminf(fmaxf(u, -12.0f), 12.0f);
    float e = __expf(2.0f * u);
    float t = (e - 1.0f) / (e + 1.0f);
#endif
    return 0.5f * h * (1.0f + t);
}

static __device__ __forceinline__ v8f wmma_bf(v16bf a, v16bf b, v8f c) {
    return __builtin_amdgcn_wmma_f32_16x16x32_bf16(
        /*neg_a=*/false, a, /*neg_b=*/false, b,
        /*c_mod=*/(short)0, c, /*reuse_a=*/false, /*reuse_b=*/false);
}

// ---------------------------------------------------------------------------
// Kernel 1: convert + transpose weights f32 -> bf16 into workspace (stays in L2)
//   W1t [HID][CIN]            (B of up-proj, N-major, contiguous K)
//   Wgt [16][HID]             (B of gate, padded to 16 "experts", rows 8..15 = 0)
//   Wdt [NE][CIN][HID]        (B of down-proj per expert, N-major, contiguous K)
// ---------------------------------------------------------------------------
__global__ void moe_cvt_weights(const float* __restrict__ W1,
                                const float* __restrict__ Wg,
                                const float* __restrict__ Wd,
                                unsigned short* __restrict__ W1t,
                                unsigned short* __restrict__ Wgt,
                                unsigned short* __restrict__ Wdt) {
    const int n1 = HID * CIN;
    const int n2 = 16 * HID;
    const int n3 = NE * CIN * HID;
    int idx = blockIdx.x * blockDim.x + threadIdx.x;
    if (idx < n1) {
        int n = idx / CIN, k = idx % CIN;
        W1t[idx] = f2bf(W1[(size_t)k * HID + n]);
    } else if (idx < n1 + n2) {
        int i = idx - n1;
        int e = i / HID, k = i % HID;
        Wgt[i] = (e < NE) ? f2bf(Wg[(size_t)k * NE + e]) : (unsigned short)0;
    } else if (idx < n1 + n2 + n3) {
        int j = idx - n1 - n2;
        int e = j / (CIN * HID);
        int r = j % (CIN * HID);
        int c = r / HID, k = r % HID;
        Wdt[j] = f2bf(Wd[((size_t)e * HID + k) * CIN + c]);
    }
}

// ---------------------------------------------------------------------------
// Kernel 2: fused MoE MLP. One 128-token tile per workgroup, 8 waves.
// ---------------------------------------------------------------------------
__global__ __launch_bounds__(256, 1)
void moe_fused(const float* __restrict__ x,
               const float* __restrict__ b1,
               const float* __restrict__ bg,
               const float* __restrict__ bd,
               const unsigned short* __restrict__ W1t,
               const unsigned short* __restrict__ Wgt,
               const unsigned short* __restrict__ Wdt,
               float* __restrict__ out) {
    extern __shared__ char smem[];
    unsigned short* Hb   = (unsigned short*)smem;                 // [TM][HS] bf16 (phase 2+)
    float*          Xf   = (float*)smem;                          // [TM][XS] f32 (phase 1, overlaid)
    float*          G    = (float*)(smem + (size_t)TM * HS * 2);  // [TM][16] gate logits
    int*            tokE = (int*)(smem + (size_t)TM * HS * 2 + TM * 16 * 4);
    int*            tokSlot = tokE + TM;
    float*          tokP    = (float*)(tokSlot + TM);
    int*            perm    = (int*)(tokP + TM);                  // [256] (<=240 used)
    int*            counts  = perm + 256;                         // [NE]
    int*            offs    = counts + NE;                        // [NE+1]
    int*            grpE    = offs + NE + 1;                      // [16]
    int*            nGrp    = grpE + 16;                          // [1]

    const int tid  = threadIdx.x;
    const int wv   = tid >> 5;
    const int lane = tid & 31;
    const int nl   = lane & 15;    // N-lane / M-lane within half-wave
    const int hi   = lane >> 4;    // half-wave select
    const int t0   = blockIdx.x * TM;

    if (tid < NE) counts[tid] = 0;

    // ---- Phase 1: stage x tile (f32) into LDS (CDNA5 async global->LDS) ----
    {
        const float* xg = x + (size_t)t0 * CIN;
#pragma unroll
        for (int j = 0; j < (TM * CIN / 4) / 256; ++j) {
            int idx4 = j * 256 + tid;
            int row  = idx4 >> 6;          // 64 float4 per row
            int col  = (idx4 & 63) << 2;
#if USE_ASYNC
            unsigned ldsa = (unsigned)(size_t)&Xf[row * XS + col];
            unsigned long long ga =
                (unsigned long long)(size_t)(xg + (size_t)row * CIN + col);
            asm volatile("global_load_async_to_lds_b128 %0, %1, off"
                         :: "v"(ldsa), "v"(ga) : "memory");
#else
            *(float4*)&Xf[row * XS + col] =
                *(const float4*)(xg + (size_t)row * CIN + col);
#endif
        }
#if USE_ASYNC
        asm volatile("s_wait_asynccnt 0" ::: "memory");
#endif
    }
    __syncthreads();

    // ---- Phase 2a: build bf16 A-fragments for this wave's 16 tokens -------
    // 16-bit A 16x32 layout: lane<16 holds K=k0+{0..7,16..23}; lane>=16 holds
    // K=k0+{8..15,24..31}; M = lane&15.
    BFV a[8];
    {
        const int row = wv * 16 + nl;
#pragma unroll
        for (int kc = 0; kc < 8; ++kc) {
            const int cA = kc * 32 + hi * 8;
            const float* p = &Xf[row * XS + cA];
            float4 f0 = *(const float4*)(p);
            float4 f1 = *(const float4*)(p + 4);
            float4 f2 = *(const float4*)(p + 16);
            float4 f3 = *(const float4*)(p + 20);
            a[kc].s[0]  = f2bf(f0.x); a[kc].s[1]  = f2bf(f0.y);
            a[kc].s[2]  = f2bf(f0.z); a[kc].s[3]  = f2bf(f0.w);
            a[kc].s[4]  = f2bf(f1.x); a[kc].s[5]  = f2bf(f1.y);
            a[kc].s[6]  = f2bf(f1.z); a[kc].s[7]  = f2bf(f1.w);
            a[kc].s[8]  = f2bf(f2.x); a[kc].s[9]  = f2bf(f2.y);
            a[kc].s[10] = f2bf(f2.z); a[kc].s[11] = f2bf(f2.w);
            a[kc].s[12] = f2bf(f3.x); a[kc].s[13] = f2bf(f3.y);
            a[kc].s[14] = f2bf(f3.z); a[kc].s[15] = f2bf(f3.w);
        }
    }
    __syncthreads();   // x fully consumed; LDS region reused for H

    // ---- Phase 2b: up-proj H = gelu(x @ W1 + b1), H -> LDS (bf16) ---------
#pragma unroll 1
    for (int n0t = 0; n0t < HID / 16; ++n0t) {
        const int n0 = n0t * 16;
        v8f acc = {};
#pragma unroll
        for (int kc = 0; kc < 8; ++kc) {
            BFV bv;
            const unsigned short* bp =
                W1t + (size_t)(n0 + nl) * CIN + kc * 32 + hi * 8;
            bv.u[0] = *(const uint4*)bp;
            bv.u[1] = *(const uint4*)(bp + 16);
            acc = wmma_bf(a[kc].v, bv.v, acc);
        }
        const float bias = b1[n0 + nl];
#pragma unroll
        for (int r = 0; r < 8; ++r) {
            float h = acc[r] + bias;
            float g = gelu_act(h);
            Hb[(wv * 16 + r + 8 * hi) * HS + n0 + nl] = f2bf(g);
        }
    }

    // ---- Phase 3: gate logits via WMMA (Wg padded to N=16), softmax/argmax -
    {
        v8f gacc = {};
        const int rowA = wv * 16 + nl;
#pragma unroll 1
        for (int kc = 0; kc < HID / 32; ++kc) {
            const int cA = kc * 32 + hi * 8;
            BFV av, bv;
            av.u[0] = *(const uint4*)&Hb[rowA * HS + cA];
            av.u[1] = *(const uint4*)&Hb[rowA * HS + cA + 16];
            const unsigned short* bp = Wgt + (size_t)nl * HID + kc * 32 + hi * 8;
            bv.u[0] = *(const uint4*)bp;
            bv.u[1] = *(const uint4*)(bp + 16);
            gacc = wmma_bf(av.v, bv.v, gacc);
        }
#pragma unroll
        for (int r = 0; r < 8; ++r)
            G[(wv * 16 + r + 8 * hi) * 16 + nl] = gacc[r];

        if (lane < 16) {           // one lane per token of this wave
            const int t = wv * 16 + lane;
            float l[NE];
            float m = -1e30f;
#pragma unroll
            for (int e = 0; e < NE; ++e) {
                l[e] = G[t * 16 + e] + bg[e];
                m = fmaxf(m, l[e]);
            }
            float s = 0.f, best = -1e30f;
            int be = 0;
#pragma unroll
            for (int e = 0; e < NE; ++e) {
                float ex = __expf(l[e] - m);
                s += ex;
                if (l[e] > best) { best = l[e]; be = e; }
            }
            tokE[t]    = be;
            tokP[t]    = __expf(best - m) / s;
            tokSlot[t] = atomicAdd(&counts[be], 1);
        }
    }
    __syncthreads();

    // ---- Phase 3b: counting-sort into expert buckets, pad to 16 rows ------
    if (tid == 0) {
        int o = 0;
#pragma unroll
        for (int e = 0; e < NE; ++e) {
            offs[e] = o;
            o += ((counts[e] + 15) >> 4) << 4;
        }
        offs[NE] = o;
        nGrp[0]  = o >> 4;
    }
    perm[tid] = -1;                     // 256 threads cover 256 perm slots
    __syncthreads();
    if (tid < TM) perm[offs[tokE[tid]] + tokSlot[tid]] = tid;
    if (tid < 16) {
        int ge = 0;
#pragma unroll
        for (int e = 0; e < NE; ++e)
            if (offs[e] <= tid * 16 && tid * 16 < offs[e + 1]) ge = e;
        grpE[tid] = ge;
    }
    __syncthreads();

    // ---- Phase 4: down-proj, expert-homogeneous 16-row groups -------------
    const int ng = nGrp[0];
#pragma unroll 1
    for (int g = wv; g < ng; g += NWAVE) {
        const int e    = grpE[g];
        const int tokA = perm[g * 16 + nl];
        const int rowH = (tokA < 0) ? 0 : tokA;   // pad rows read junk, masked on store
        const unsigned short* Wde = Wdt + (size_t)e * CIN * HID;

        v8f zero = {};
        v8f acc[16];                              // all 16 N-tiles of C=256
#pragma unroll
        for (int i = 0; i < 16; ++i) acc[i] = zero;

#pragma unroll 1
        for (int kc = 0; kc < HID / 32; ++kc) {
            const int cA = kc * 32 + hi * 8;
            BFV av;
            av.u[0] = *(const uint4*)&Hb[rowH * HS + cA];
            av.u[1] = *(const uint4*)&Hb[rowH * HS + cA + 16];
            const int koff = kc * 32 + hi * 8;
#pragma unroll
            for (int nt = 0; nt < 16; ++nt) {
                BFV bv;
                const unsigned short* bp = Wde + (size_t)(nt * 16 + nl) * HID + koff;
                bv.u[0] = *(const uint4*)bp;
                bv.u[1] = *(const uint4*)(bp + 16);
                acc[nt] = wmma_bf(av.v, bv.v, acc[nt]);
            }
        }

#pragma unroll
        for (int r = 0; r < 8; ++r) {
            const int tokR = perm[g * 16 + r + 8 * hi];
            if (tokR >= 0) {
                const float p  = tokP[tokR];
                float*      op = out + (size_t)(t0 + tokR) * CIN;
#pragma unroll
                for (int nt = 0; nt < 16; ++nt) {
                    const int n = nt * 16 + nl;
                    op[n] = p * (acc[nt][r] + bd[e * CIN + n]);
                }
            }
        }
    }
}

extern "C" void kernel_launch(void* const* d_in, const int* in_sizes, int n_in,
                              void* d_out, int out_size, void* d_ws, size_t ws_size,
                              hipStream_t stream) {
    const float* x  = (const float*)d_in[0];
    const float* W1 = (const float*)d_in[1];
    const float* b1 = (const float*)d_in[2];
    const float* Wg = (const float*)d_in[3];
    const float* bg = (const float*)d_in[4];
    const float* Wd = (const float*)d_in[5];
    const float* bd = (const float*)d_in[6];
    float*       out = (float*)d_out;

    unsigned short* W1t = (unsigned short*)d_ws;
    unsigned short* Wgt = W1t + HID * CIN;
    unsigned short* Wdt = Wgt + 16 * HID;

    const int T = in_sizes[0] / CIN;

    const int cvt_total = HID * CIN + 16 * HID + NE * CIN * HID;
    moe_cvt_weights<<<(cvt_total + 255) / 256, 256, 0, stream>>>(
        W1, Wg, Wd, W1t, Wgt, Wdt);

    const size_t lds_bytes = (size_t)TM * HS * 2      // H (bf16)
                           + (size_t)TM * 16 * 4      // gate logits
                           + (size_t)TM * 4 * 3       // tokE / tokSlot / tokP
                           + 256 * 4                  // perm
                           + (NE + NE + 1 + 16 + 1) * 4;
    (void)hipFuncSetAttribute((const void*)moe_fused,
                              hipFuncAttributeMaxDynamicSharedMemorySize,
                              (int)lds_bytes);
    moe_fused<<<T / TM, 256, lds_bytes, stream>>>(
        x, b1, bg, bd, W1t, Wgt, Wdt, out);
}
